// TransLayer_670014898671
// MI455X (gfx1250) — matrix operand order repaired and checked by reading, not compiled, not run on hardware
//
#include <hip/hip_runtime.h>
#include <hip/hip_bf16.h>
#include <cstdint>

typedef _Float16 HH;
typedef __attribute__((ext_vector_type(16))) _Float16 v16h;
typedef __attribute__((ext_vector_type(8)))  _Float16 v8h;
typedef __attribute__((ext_vector_type(8)))  float    v8f;
typedef __attribute__((ext_vector_type(2)))  float    v2f;

#define DIMD   512
#define HEADS  8
#define DHEAD  64
#define INNERD 512
#define MLAND  256
#define PITERS 6
#define KCONV  33
#define BB     2
#define NSEQ   4096
#define QSCALE 0.125f
#define LN_EPS 1e-5f

// ---------------- CDNA5 async-copy helpers (gfx1250 ASYNCcnt path) ----------

// one 16-byte global->LDS async copy per lane (VGLOBAL opcode 98)
__device__ inline void async_copy_b128(uint32_t lds_off, const void* gsrc) {
  asm volatile("global_load_async_to_lds_b128 %0, %1, off"
               :: "v"(lds_off), "v"(gsrc) : "memory");
}
__device__ inline void wait_asynccnt0() {
  asm volatile("s_wait_asynccnt 0x0" ::: "memory");
}

// ---------------- WMMA fragment helpers (CDNA5 wave32 layouts) ----------------

// 16-bit A matrix 16x32: lane holds row M=lane&15; kb=(lane>>4)*8;
// halves 0..7 -> K=kb..kb+7, halves 8..15 -> K=16+kb..16+kb+7
__device__ inline v16h load_a16(const HH* __restrict__ A, int lda, int k0) {
  int lane = threadIdx.x & 31;
  int row  = lane & 15;
  int kb   = (lane >> 4) << 3;
  union { v16h v; v8h h[2]; } u;
  u.h[0] = *(const v8h*)(A + (size_t)row * lda + k0 + kb);
  u.h[1] = *(const v8h*)(A + (size_t)row * lda + k0 + 16 + kb);
  return u.v;
}

// B fragment from an LDS-staged [32 rows x 64 halves] panel:
// lane col n=lane&15 (+rowOff for second tile); substep s in {0,32};
// kb=(lane>>4)*8; halves 0..7 -> K=kb..kb+7, halves 8..15 -> K=16+kb..16+kb+7
__device__ inline v16h load_b16_lds(const HH* sB, int rowOff, int s) {
  int lane = threadIdx.x & 31;
  int n    = (lane & 15) + rowOff;
  int kb   = (lane >> 4) << 3;
  union { v16h v; v8h h[2]; } u;
  u.h[0] = *(const v8h*)(sB + (size_t)n * 64 + s + kb);
  u.h[1] = *(const v8h*)(sB + (size_t)n * 64 + s + 16 + kb);
  return u.v;
}

// f32 C/D 16x16: VGPR r -> row=(lane>>4)*8+r, col=lane&15
__device__ inline void store_c(float* __restrict__ C, int ldc, const v8f& c) {
  int lane = threadIdx.x & 31;
  int col  = lane & 15;
  int rb   = (lane >> 4) << 3;
#pragma unroll
  for (int r = 0; r < 8; ++r) C[(size_t)(rb + r) * ldc + col] = c[r];
}

// transposed store: element (row,col) of the tile -> C[col*ldc + row]
__device__ inline void store_ct(float* __restrict__ C, int ldc, const v8f& c) {
  int lane = threadIdx.x & 31;
  int col  = lane & 15;
  int rb   = (lane >> 4) << 3;
#pragma unroll
  for (int r = 0; r < 8; ++r) C[(size_t)col * ldc + rb + r] = c[r];
}

// ---------------- GEMM kernels ----------------

// Batched f16 NT GEMM, fp32 accumulate.
// Block = 4 waves; each wave owns a 32-row M tile (2x2 register blocking),
// all waves share one 32-col B panel, double-buffered in LDS via async copies
// (copy of chunk i+1 overlaps the 8 WMMAs of chunk i). K % 64 == 0.
// C[M,N] = alpha * A[M,K] * B[N,K]^T ; CT=true stores C transposed as [N,M].
template <bool CT>
__global__ __launch_bounds__(128) void gemm16_kernel(
    const HH* __restrict__ A, const HH* __restrict__ B, float* __restrict__ C,
    int M, int N, int K, long sA, long sB_, long sC, float alpha) {
  __shared__ HH sB[2][32 * 64];         // two 4 KB k-chunks of the B panel

  int bt = blockIdx.z;
  const HH* Ab = A + (size_t)bt * sA;
  const HH* Bb = B + (size_t)bt * sB_;
  float*    Cb = C + (size_t)bt * sC;

  int tn  = blockIdx.x;                                   // 32-col tile (uniform in block)
  int tmw = blockIdx.y * 4 + (threadIdx.x >> 5);          // this wave's 32-row tile
  bool active = (tmw * 32 < M);                           // wave-uniform
  int  tm  = active ? tmw : 0;

  const HH* At0 = Ab + (size_t)tm * 32 * K;
  const HH* At1 = At0 + (size_t)16 * K;
  const HH* Bt  = Bb + (size_t)tn * 32 * K;

  // per-thread async-copy slots: 128 threads x 2x16B = 32 rows x 128B
  int crow = threadIdx.x >> 2;          // 0..31  (B panel row)
  int cchk = threadIdx.x & 3;           // 0..3   (16-byte chunk within 64B half-row)
  uint32_t lds0 = (uint32_t)(uintptr_t)(&sB[0][0]) + (uint32_t)(crow * 128 + cchk * 16);
  uint32_t lds1 = (uint32_t)(uintptr_t)(&sB[1][0]) + (uint32_t)(crow * 128 + cchk * 16);
  const HH* gsrc = Bt + (size_t)crow * K + cchk * 8;

  int lane = threadIdx.x & 31;
  const HH* pfA = At0 + (size_t)(lane & 15) * K + ((lane >> 4) << 3);

  // prologue: stage chunk 0 into buffer 0
  async_copy_b128(lds0, (const void*)gsrc);
  async_copy_b128(lds0 + 64, (const void*)(gsrc + 32));

  v8f a00 = {}, a01 = {}, a10 = {}, a11 = {};
  int nchunks = K >> 6;
  for (int it = 0; it < nchunks; ++it) {
    int k0 = it << 6;
    const HH* sBc   = &sB[it & 1][0];
    uint32_t ldsNxt = (it & 1) ? lds0 : lds1;
    wait_asynccnt0();                   // our copies of current chunk landed
    __syncthreads();                    // everyone's copies landed; prev reads done
    if (it + 1 < nchunks) {             // overlap next chunk's DMA with compute
      async_copy_b128(ldsNxt, (const void*)(gsrc + k0 + 64));
      async_copy_b128(ldsNxt + 64, (const void*)(gsrc + k0 + 96));
    }
    if (k0 + 256 < K) {                 // gfx1250 global_prefetch_b8 for A stream
      __builtin_prefetch((const void*)(pfA + k0 + 256), 0, 1);
    }
#pragma unroll
    for (int s = 0; s < 64; s += 32) {
      v16h fa0 = load_a16(At0, K, k0 + s);
      v16h fa1 = load_a16(At1, K, k0 + s);
      v16h fb0 = load_b16_lds(sBc, 0, s);
      v16h fb1 = load_b16_lds(sBc, 16, s);
      a00 = __builtin_amdgcn_wmma_f32_16x16x32_f16(false, fa0, false, fb0, (short)0, a00, false, false);
      a01 = __builtin_amdgcn_wmma_f32_16x16x32_f16(false, fa0, false, fb1, (short)0, a01, false, false);
      a10 = __builtin_amdgcn_wmma_f32_16x16x32_f16(false, fa1, false, fb0, (short)0, a10, false, false);
      a11 = __builtin_amdgcn_wmma_f32_16x16x32_f16(false, fa1, false, fb1, (short)0, a11, false, false);
    }
  }
  if (!active) return;                  // wave-uniform; barriers all done
  if (alpha != 1.0f) {
#pragma unroll
    for (int r = 0; r < 8; ++r) { a00[r] *= alpha; a01[r] *= alpha; a10[r] *= alpha; a11[r] *= alpha; }
  }
  if (!CT) {
    float* Ct = Cb + (size_t)tm * 32 * N + tn * 32;
    store_c(Ct, N, a00);
    store_c(Ct + 16, N, a01);
    store_c(Ct + (size_t)16 * N, N, a10);
    store_c(Ct + (size_t)16 * N + 16, N, a11);
  } else {                              // output [N,M] row-major
    float* Ct = Cb + (size_t)tn * 32 * M + tm * 32;
    store_ct(Ct, M, a00);
    store_ct(Ct + (size_t)16 * M, M, a01);
    store_ct(Ct + 16, M, a10);
    store_ct(Ct + (size_t)16 * M + 16, M, a11);
  }
}

// Batched fp32 NN GEMM via v_wmma_f32_16x16x4_f32, 2x2 register-blocked,
// for the numerically sensitive Newton-Schulz pinv iteration.
// C = alpha * A[M,K] * B[K,N]
__global__ __launch_bounds__(128) void gemm32nn_kernel(
    const float* __restrict__ A, const float* __restrict__ B, float* __restrict__ C,
    int M, int N, int K, long sA, long sB, long sC, float alpha) {
  int bt = blockIdx.z;
  const float* Ab = A + (size_t)bt * sA;
  const float* Bb = B + (size_t)bt * sB;
  float*       Cb = C + (size_t)bt * sC;
  int tn = blockIdx.x * (blockDim.x >> 5) + (threadIdx.x >> 5);
  int tm = blockIdx.y;
  if (tn * 32 >= N || tm * 32 >= M) return;
  int lane = threadIdx.x & 31;
  int mi   = lane & 15;
  int kb   = (lane >> 4) << 1;        // f32 A 16x4: lanes<16 K=0,1; lanes>=16 K=2,3
  const float* Ar0 = Ab + (size_t)(tm * 32 + mi) * K;
  const float* Ar1 = Ar0 + (size_t)16 * K;
  const float* Bc0 = Bb + tn * 32 + mi;
  const float* Bc1 = Bc0 + 16;
  v8f a00 = {}, a01 = {}, a10 = {}, a11 = {};
  for (int k0 = 0; k0 < K; k0 += 4) {
    v2f fa0 = {Ar0[k0 + kb], Ar0[k0 + kb + 1]};
    v2f fa1 = {Ar1[k0 + kb], Ar1[k0 + kb + 1]};
    v2f fb0 = {Bc0[(size_t)(k0 + kb) * N], Bc0[(size_t)(k0 + kb + 1) * N]};
    v2f fb1 = {Bc1[(size_t)(k0 + kb) * N], Bc1[(size_t)(k0 + kb + 1) * N]};
    a00 = __builtin_amdgcn_wmma_f32_16x16x4_f32(false, fa0, false, fb0, (short)0, a00, false, false);
    a01 = __builtin_amdgcn_wmma_f32_16x16x4_f32(false, fa0, false, fb1, (short)0, a01, false, false);
    a10 = __builtin_amdgcn_wmma_f32_16x16x4_f32(false, fa1, false, fb0, (short)0, a10, false, false);
    a11 = __builtin_amdgcn_wmma_f32_16x16x4_f32(false, fa1, false, fb1, (short)0, a11, false, false);
  }
  if (alpha != 1.0f) {
#pragma unroll
    for (int r = 0; r < 8; ++r) { a00[r] *= alpha; a01[r] *= alpha; a10[r] *= alpha; a11[r] *= alpha; }
  }
  float* Ct = Cb + (size_t)tm * 32 * N + tn * 32;
  store_c(Ct, N, a00);
  store_c(Ct + 16, N, a01);
  store_c(Ct + (size_t)16 * N, N, a10);
  store_c(Ct + (size_t)16 * N + 16, N, a11);
}

// ---------------- elementwise / reduction kernels ----------------

__global__ void ln_kernel(const float* __restrict__ x, const float* __restrict__ w,
                          const float* __restrict__ b, HH* __restrict__ xn) {
  int row = blockIdx.x;
  const float* xr = x + (size_t)row * DIMD;
  __shared__ float red[256];
  int t = threadIdx.x;
  float v0 = xr[t], v1 = xr[t + 256];
  red[t] = v0 + v1;
  __syncthreads();
  for (int o = 128; o > 0; o >>= 1) { if (t < o) red[t] += red[t + o]; __syncthreads(); }
  float mu = red[0] * (1.0f / DIMD);
  __syncthreads();
  float d0 = v0 - mu, d1 = v1 - mu;
  red[t] = d0 * d0 + d1 * d1;
  __syncthreads();
  for (int o = 128; o > 0; o >>= 1) { if (t < o) red[t] += red[t + o]; __syncthreads(); }
  float rs = rsqrtf(red[0] * (1.0f / DIMD) + LN_EPS);
  HH* o = xn + (size_t)row * DIMD;
  o[t]       = (HH)(d0 * rs * w[t] + b[t]);
  o[t + 256] = (HH)(d1 * rs * w[t + 256] + b[t + 256]);
}

__global__ void cvt_h_kernel(const float* __restrict__ s, HH* __restrict__ d, long n) {
  long i = (long)blockIdx.x * blockDim.x + threadIdx.x;
  if (i < n) d[i] = (HH)s[i];
}

__global__ void split_kernel(const float* __restrict__ qkv, HH* __restrict__ q,
                             HH* __restrict__ k, HH* __restrict__ v) {
  size_t i = (size_t)blockIdx.x * blockDim.x + threadIdx.x;   // (b,n,h,d)
  if (i >= (size_t)BB * NSEQ * INNERD) return;
  int d = i & 63;
  int h = (i >> 6) & 7;
  int n = (i >> 9) & (NSEQ - 1);
  int b = i >> 21;
  const float* src = qkv + (size_t)(b * NSEQ + n) * (3 * INNERD) + h * DHEAD + d;
  size_t dst = (((size_t)b * HEADS + h) * NSEQ + n) * DHEAD + d;
  q[dst] = (HH)(src[0] * QSCALE);
  k[dst] = (HH)(src[INNERD]);
  v[dst] = (HH)(src[2 * INNERD]);
}

// [BH, NSEQ, DHEAD] -> [BH, DHEAD, NSEQ] (f16)
__global__ void transpose_h_kernel(const HH* __restrict__ src, HH* __restrict__ dst) {
  size_t i = (size_t)blockIdx.x * blockDim.x + threadIdx.x;   // (bh,d,n)
  if (i >= (size_t)BB * HEADS * DHEAD * NSEQ) return;
  int n = i & (NSEQ - 1);
  int d = (i >> 12) & (DHEAD - 1);
  size_t bh = i >> 18;
  dst[i] = src[(bh * NSEQ + (size_t)n) * DHEAD + d];
}

__global__ void land_kernel(const HH* __restrict__ t, HH* __restrict__ land) {
  size_t i = (size_t)blockIdx.x * blockDim.x + threadIdx.x;   // (bh,m,d)
  if (i >= (size_t)BB * HEADS * MLAND * DHEAD) return;
  int d = i & 63;
  int m = (i >> 6) & (MLAND - 1);
  size_t bh = i >> 14;
  const HH* src = t + (bh * NSEQ + (size_t)m * 16) * DHEAD + d;
  float s = 0.f;
#pragma unroll
  for (int l = 0; l < 16; ++l) s += (float)src[(size_t)l * DHEAD];
  land[i] = (HH)(s * (1.0f / 16.0f));
}

// row softmax; outh (f16) and/or outf (f32) may be null
__global__ void softmax_kernel(const float* __restrict__ S, HH* __restrict__ outh,
                               float* __restrict__ outf, int W) {
  int row = blockIdx.x;
  const float* sr = S + (size_t)row * W;
  __shared__ float red[256];
  int t = threadIdx.x;
  float m = -1e30f;
  for (int i = t; i < W; i += 256) m = fmaxf(m, sr[i]);
  red[t] = m; __syncthreads();
  for (int o = 128; o > 0; o >>= 1) { if (t < o) red[t] = fmaxf(red[t], red[t + o]); __syncthreads(); }
  float rm = red[0]; __syncthreads();
  float s = 0.f;
  for (int i = t; i < W; i += 256) s += __expf(sr[i] - rm);
  red[t] = s; __syncthreads();
  for (int o = 128; o > 0; o >>= 1) { if (t < o) red[t] += red[t + o]; __syncthreads(); }
  float inv = 1.0f / red[0];
  for (int i = t; i < W; i += 256) {
    float e = __expf(sr[i] - rm) * inv;
    if (outh) outh[(size_t)row * W + i] = (HH)e;
    if (outf) outf[(size_t)row * W + i] = e;
  }
}

__global__ void zero2_kernel(float* p) { if (threadIdx.x < 2) p[threadIdx.x] = 0.0f; }

__global__ void rowsum_max_kernel(const float* __restrict__ a, float* pmax) {
  int row = blockIdx.x;                          // global row over BH*M
  __shared__ float red[256];
  int t = threadIdx.x;
  red[t] = fabsf(a[(size_t)row * MLAND + t]);
  __syncthreads();
  for (int o = 128; o > 0; o >>= 1) { if (t < o) red[t] += red[t + o]; __syncthreads(); }
  if (t == 0) atomicMax((int*)pmax, __float_as_int(red[0]));
}

__global__ void colsum_max_kernel(const float* __restrict__ a, float* pmax) {
  int c  = blockIdx.x & (MLAND - 1);
  int bh = blockIdx.x >> 8;
  __shared__ float red[256];
  int t = threadIdx.x;
  red[t] = fabsf(a[((size_t)bh * MLAND + t) * MLAND + c]);
  __syncthreads();
  for (int o = 128; o > 0; o >>= 1) { if (t < o) red[t] += red[t + o]; __syncthreads(); }
  if (t == 0) atomicMax((int*)pmax, __float_as_int(red[0]));
}

__global__ void zt_init_kernel(const float* __restrict__ a, const float* __restrict__ mx,
                               float* __restrict__ z) {
  size_t i = (size_t)blockIdx.x * blockDim.x + threadIdx.x;
  if (i >= (size_t)BB * HEADS * MLAND * MLAND) return;
  int c = i & (MLAND - 1);
  int r = (i >> 8) & (MLAND - 1);
  size_t bh = i >> 16;
  float inv = 1.0f / (mx[0] * mx[1]);
  z[i] = a[((size_t)bh * MLAND + c) * MLAND + r] * inv;   // transpose
}

// dst = a*src + bI*I  (batched 256x256)
__global__ void axpb_ident_kernel(const float* __restrict__ s, float* __restrict__ d,
                                  float a, float bI) {
  size_t i = (size_t)blockIdx.x * blockDim.x + threadIdx.x;
  if (i >= (size_t)BB * HEADS * MLAND * MLAND) return;
  int c = i & (MLAND - 1);
  int r = (i >> 8) & (MLAND - 1);
  d[i] = a * s[i] + ((r == c) ? bI : 0.0f);
}

// depthwise residual conv over sequence, accumulate into out_heads (f32)
__global__ void conv_add_kernel(const HH* __restrict__ v, const float* __restrict__ cw,
                                float* __restrict__ oh) {
  size_t i = (size_t)blockIdx.x * blockDim.x + threadIdx.x;   // (bh,n,d)
  if (i >= (size_t)BB * HEADS * NSEQ * DHEAD) return;
  int d = i & 63;
  int n = (i >> 6) & (NSEQ - 1);
  size_t bh = i >> 18;
  int h = bh & (HEADS - 1);
  const HH* vb = v + bh * (size_t)NSEQ * DHEAD + d;
  float s = 0.f;
#pragma unroll
  for (int kk = 0; kk < KCONV; ++kk) {
    int nn = n + kk - KCONV / 2;
    if (nn >= 0 && nn < NSEQ) s += (float)vb[(size_t)nn * DHEAD] * cw[h * KCONV + kk];
  }
  oh[i] += s;
}

__global__ void concat_kernel(const float* __restrict__ oh, HH* __restrict__ attn) {
  size_t i = (size_t)blockIdx.x * blockDim.x + threadIdx.x;   // (b,n,h,d)
  if (i >= (size_t)BB * NSEQ * INNERD) return;
  int d = i & 63;
  int h = (i >> 6) & 7;
  int n = (i >> 9) & (NSEQ - 1);
  int b = i >> 21;
  attn[i] = (HH)oh[(((size_t)b * HEADS + h) * NSEQ + n) * DHEAD + d];
}

__global__ void final_kernel(const float* __restrict__ x, const float* __restrict__ proj,
                             const float* __restrict__ ob, float* __restrict__ out) {
  size_t i = (size_t)blockIdx.x * blockDim.x + threadIdx.x;
  if (i >= (size_t)BB * NSEQ * DIMD) return;
  out[i] = x[i] + proj[i] + ob[i & (DIMD - 1)];
}

// ---------------- host launcher ----------------

static inline dim3 gemm16_grid(int M, int N, int batches) {
  return dim3(N / 32, (M + 127) / 128, batches);
}
static inline dim3 gemm32_grid(int M, int N, int batches) {
  return dim3((N / 32 + 3) / 4, M / 32, batches);
}

extern "C" void kernel_launch(void* const* d_in, const int* in_sizes, int n_in,
                              void* d_out, int out_size, void* d_ws, size_t ws_size,
                              hipStream_t stream) {
  const float* x      = (const float*)d_in[0];
  const float* ln_w   = (const float*)d_in[1];
  const float* ln_b   = (const float*)d_in[2];
  const float* qkv_w  = (const float*)d_in[3];
  const float* out_w  = (const float*)d_in[4];
  const float* out_b  = (const float*)d_in[5];
  const float* conv_w = (const float*)d_in[6];
  float* out = (float*)d_out;

  char* base = (char*)d_ws;
  size_t off = 0;
  auto alloc = [&](size_t bytes) -> char* {
    char* p = base + off;
    off = (off + bytes + 255) & ~(size_t)255;
    return p;
  };

  const size_t BN = (size_t)BB * NSEQ;          // 8192
  const int    BH = BB * HEADS;                 // 16

  // 64 MiB region shared by qkv_f (48 MiB) then sim_f (64 MiB)
  float* big0     = (float*)alloc((size_t)BH * NSEQ * MLAND * 4);
  float* qkv_f    = big0;
  float* sim_f    = big0;
  HH*    a1_h     = (HH*)alloc((size_t)BH * NSEQ * MLAND * 2);
  HH*    a3_h     = (HH*)alloc((size_t)BH * MLAND * NSEQ * 2);
  HH*    xn_h     = (HH*)alloc(BN * DIMD * 2);
  HH*    qkvw_h   = (HH*)alloc((size_t)3 * INNERD * DIMD * 2);
  HH*    outw_h   = (HH*)alloc((size_t)DIMD * INNERD * 2);
  HH*    q_h      = (HH*)alloc((size_t)BH * NSEQ * DHEAD * 2);
  HH*    k_h      = (HH*)alloc((size_t)BH * NSEQ * DHEAD * 2);
  HH*    v_h      = (HH*)alloc((size_t)BH * NSEQ * DHEAD * 2);
  HH*    vT_h     = (HH*)alloc((size_t)BH * DHEAD * NSEQ * 2);
  HH*    qland_h  = (HH*)alloc((size_t)BH * MLAND * DHEAD * 2);
  HH*    kland_h  = (HH*)alloc((size_t)BH * MLAND * DHEAD * 2);
  float* sim2_f   = (float*)alloc((size_t)BH * MLAND * MLAND * 4);
  float* a2_f     = (float*)alloc((size_t)BH * MLAND * MLAND * 4);
  float* z0       = (float*)alloc((size_t)BH * MLAND * MLAND * 4);
  float* z1       = (float*)alloc((size_t)BH * MLAND * MLAND * 4);
  float* xz_f     = (float*)alloc((size_t)BH * MLAND * MLAND * 4);
  float* t_f      = (float*)alloc((size_t)BH * MLAND * MLAND * 4);
  float* u_f      = (float*)alloc((size_t)BH * MLAND * MLAND * 4);
  float* w_f      = (float*)alloc((size_t)BH * MLAND * MLAND * 4);
  HH*    a2inv_h  = (HH*)alloc((size_t)BH * MLAND * MLAND * 2);
  float* a3vT_f   = (float*)alloc((size_t)BH * DHEAD * MLAND * 4);   // [64,256] per bh
  HH*    a3vT_h   = (HH*)alloc((size_t)BH * DHEAD * MLAND * 2);
  float* wsmT_f   = (float*)alloc((size_t)BH * DHEAD * MLAND * 4);   // [64,256] per bh
  HH*    wsmT_h   = (HH*)alloc((size_t)BH * DHEAD * MLAND * 2);
  float* oh_f     = (float*)alloc((size_t)BH * NSEQ * DHEAD * 4);
  HH*    attn_h   = (HH*)alloc(BN * INNERD * 2);
  float* proj_f   = (float*)alloc(BN * DIMD * 4);
  float* maxes    = (float*)alloc(256);
  (void)ws_size; (void)in_sizes; (void)n_in; (void)out_size;

  // weight conversion
  cvt_h_kernel<<<(3 * INNERD * DIMD + 255) / 256, 256, 0, stream>>>(qkv_w, qkvw_h, (long)3 * INNERD * DIMD);
  cvt_h_kernel<<<(DIMD * INNERD + 255) / 256, 256, 0, stream>>>(out_w, outw_h, (long)DIMD * INNERD);

  // LayerNorm
  ln_kernel<<<(int)BN, 256, 0, stream>>>(x, ln_w, ln_b, xn_h);

  // QKV projection: [8192,512] x [1536,512]^T
  gemm16_kernel<false><<<gemm16_grid((int)BN, 3 * INNERD, 1), 128, 0, stream>>>(
      xn_h, qkvw_h, qkv_f, (int)BN, 3 * INNERD, DIMD, 0, 0, 0, 1.0f);

  split_kernel<<<(int)(BN * INNERD / 256), 256, 0, stream>>>(qkv_f, q_h, k_h, v_h);
  transpose_h_kernel<<<(int)((size_t)BH * DHEAD * NSEQ / 256), 256, 0, stream>>>(v_h, vT_h);
  land_kernel<<<(BH * MLAND * DHEAD) / 256, 256, 0, stream>>>(q_h, qland_h);
  land_kernel<<<(BH * MLAND * DHEAD) / 256, 256, 0, stream>>>(k_h, kland_h);

  // sim1 = q @ k_land^T
  gemm16_kernel<false><<<gemm16_grid(NSEQ, MLAND, BH), 128, 0, stream>>>(
      q_h, kland_h, sim_f, NSEQ, MLAND, DHEAD,
      (long)NSEQ * DHEAD, (long)MLAND * DHEAD, (long)NSEQ * MLAND, 1.0f);
  softmax_kernel<<<BH * NSEQ, 256, 0, stream>>>(sim_f, a1_h, nullptr, MLAND);

  // sim3 = q_land @ k^T  (reuses sim_f)
  gemm16_kernel<false><<<gemm16_grid(MLAND, NSEQ, BH), 128, 0, stream>>>(
      qland_h, k_h, sim_f, MLAND, NSEQ, DHEAD,
      (long)MLAND * DHEAD, (long)NSEQ * DHEAD, (long)MLAND * NSEQ, 1.0f);
  softmax_kernel<<<BH * MLAND, 256, 0, stream>>>(sim_f, a3_h, nullptr, NSEQ);

  // sim2 = q_land @ k_land^T
  gemm16_kernel<false><<<gemm16_grid(MLAND, MLAND, BH), 128, 0, stream>>>(
      qland_h, kland_h, sim2_f, MLAND, MLAND, DHEAD,
      (long)MLAND * DHEAD, (long)MLAND * DHEAD, (long)MLAND * MLAND, 1.0f);
  softmax_kernel<<<BH * MLAND, 256, 0, stream>>>(sim2_f, nullptr, a2_f, MLAND);

  // Moore-Penrose pinv init: z = a2^T / (max rowsum * max colsum)
  zero2_kernel<<<1, 32, 0, stream>>>(maxes);
  rowsum_max_kernel<<<BH * MLAND, 256, 0, stream>>>(a2_f, maxes + 0);
  colsum_max_kernel<<<BH * MLAND, 256, 0, stream>>>(a2_f, maxes + 1);
  const int pinvElems = BH * MLAND * MLAND;
  zt_init_kernel<<<pinvElems / 256, 256, 0, stream>>>(a2_f, maxes, z0);

  // Newton-Schulz iterations (fp32 WMMA, 16x16x4)
  long sP = (long)MLAND * MLAND;
  dim3 gP = gemm32_grid(MLAND, MLAND, BH);
  float* zc = z0; float* zn = z1;
  for (int it = 0; it < PITERS; ++it) {
    gemm32nn_kernel<<<gP, 128, 0, stream>>>(a2_f, zc, xz_f, MLAND, MLAND, MLAND, sP, sP, sP, 1.0f);
    axpb_ident_kernel<<<pinvElems / 256, 256, 0, stream>>>(xz_f, t_f, -1.0f, 7.0f);
    gemm32nn_kernel<<<gP, 128, 0, stream>>>(xz_f, t_f, u_f, MLAND, MLAND, MLAND, sP, sP, sP, 1.0f);
    axpb_ident_kernel<<<pinvElems / 256, 256, 0, stream>>>(u_f, t_f, -1.0f, 15.0f);
    gemm32nn_kernel<<<gP, 128, 0, stream>>>(xz_f, t_f, w_f, MLAND, MLAND, MLAND, sP, sP, sP, 1.0f);
    axpb_ident_kernel<<<pinvElems / 256, 256, 0, stream>>>(w_f, t_f, -1.0f, 13.0f);
    gemm32nn_kernel<<<gP, 128, 0, stream>>>(zc, t_f, zn, MLAND, MLAND, MLAND, sP, sP, sP, 0.25f);
    float* tmp = zc; zc = zn; zn = tmp;
  }
  cvt_h_kernel<<<pinvElems / 256, 256, 0, stream>>>(zc, a2inv_h, pinvElems);

  // a3v^T = v^T @ a3^T : NT([64,4096], [256,4096]) -> [64,256]
  gemm16_kernel<false><<<gemm16_grid(DHEAD, MLAND, BH), 128, 0, stream>>>(
      vT_h, a3_h, a3vT_f, DHEAD, MLAND, NSEQ,
      (long)DHEAD * NSEQ, (long)MLAND * NSEQ, (long)DHEAD * MLAND, 1.0f);
  cvt_h_kernel<<<(BH * DHEAD * MLAND) / 256, 256, 0, stream>>>(a3vT_f, a3vT_h, (long)BH * DHEAD * MLAND);

  // wsm = a2inv @ a3v : NT([256,256], [64,256]) -> stored transposed [64,256]
  gemm16_kernel<true><<<gemm16_grid(MLAND, DHEAD, BH), 128, 0, stream>>>(
      a2inv_h, a3vT_h, wsmT_f, MLAND, DHEAD, MLAND,
      (long)MLAND * MLAND, (long)DHEAD * MLAND, (long)DHEAD * MLAND, 1.0f);
  cvt_h_kernel<<<(BH * DHEAD * MLAND) / 256, 256, 0, stream>>>(wsmT_f, wsmT_h, (long)BH * DHEAD * MLAND);

  // out_heads = a1 @ wsm : NT([4096,256], [64,256]) -> [4096,64]
  gemm16_kernel<false><<<gemm16_grid(NSEQ, DHEAD, BH), 128, 0, stream>>>(
      a1_h, wsmT_h, oh_f, NSEQ, DHEAD, MLAND,
      (long)NSEQ * MLAND, (long)DHEAD * MLAND, (long)NSEQ * DHEAD, 1.0f);

  // depthwise conv residual on v, accumulated into out_heads
  conv_add_kernel<<<(int)((size_t)BH * NSEQ * DHEAD / 256), 256, 0, stream>>>(v_h, conv_w, oh_f);

  // concat heads -> [B,N,512] f16
  concat_kernel<<<(int)(BN * INNERD / 256), 256, 0, stream>>>(oh_f, attn_h);

  // output projection: [8192,512] x [512,512]^T
  gemm16_kernel<false><<<gemm16_grid((int)BN, DIMD, 1), 128, 0, stream>>>(
      attn_h, outw_h, proj_f, (int)BN, DIMD, INNERD, 0, 0, 0, 1.0f);

  // residual + bias
  final_kernel<<<(int)(BN * DIMD / 256), 256, 0, stream>>>(x, proj_f, out_b, out);
}
